// textPCNN_27384711479620
// MI455X (gfx1250) — matrix operand order repaired and checked by reading, not compiled, not run on hardware
//
#include <hip/hip_runtime.h>
#include <hip/hip_bf16.h>
#include <math.h>

// ---------------- problem constants ----------------
#define PN   2048          // sentences
#define PL   100           // tokens per sentence
#define PFS  3             // filter size
#define PDW  50            // word dim
#define PDPF 5             // position dim
#define PD   60            // D = 50 + 2*5
#define PDP  64            // padded D
#define PKK  192           // K = FS * PDP = 6 * 32
#define PF   230           // filters
#define PFP  240           // padded filters (15 tiles of 16)
#define FTW  5             // f-tiles per wave (register blocking)
#define PLC  98            // L - FS + 1
#define PV   100000
#define PP   200
#define PC   53
#define PBAG 8
#define PSB  2
#define NBAGS (PN / PBAG)      // 256
#define NSB   (NBAGS / PSB)    // 128
#define F3   (3 * PF)          // 690

typedef __attribute__((ext_vector_type(16))) _Float16 v16h;
typedef __attribute__((ext_vector_type(8)))  _Float16 v8h;
typedef __attribute__((ext_vector_type(8)))  float    v8f;

// Order-preserving float -> uint key for atomicMax pooling
__device__ __forceinline__ unsigned int fkey(float x) {
    unsigned int b = __float_as_uint(x);
    return (b & 0x80000000u) ? ~b : (b | 0x80000000u);
}
__device__ __forceinline__ float fdec(unsigned int k) {
    unsigned int b = (k & 0x80000000u) ? (k ^ 0x80000000u) : ~k;
    return __uint_as_float(b);
}

// ---------------- stage 1: embedding gather -> fp16 Xe[N][L][64] ----------------
__global__ __launch_bounds__(256) void gather_kernel(
    const int* __restrict__ tokens, const int* __restrict__ pf1,
    const int* __restrict__ pf2,    const float* __restrict__ wordvec,
    const float* __restrict__ pf1_emb, const float* __restrict__ pf2_emb,
    _Float16* __restrict__ Xe)
{
    int idx = blockIdx.x * blockDim.x + threadIdx.x;
    const int total = PN * PL * PDP;
    if (idx >= total) return;
    int d  = idx & (PDP - 1);
    int nl = idx >> 6;
    float v = 0.0f;
    if (d < PDW)            v = wordvec[(size_t)tokens[nl] * PDW + d];
    else if (d < PDW + PDPF) v = pf1_emb[pf1[nl] * PDPF + (d - PDW)];
    else if (d < PD)         v = pf2_emb[pf2[nl] * PDPF + (d - PDW - PDPF)];
    Xe[idx] = (_Float16)v;
}

// ---------------- stage 2: weights -> fp16 Whk[240][192] (K-major) ----------------
__global__ __launch_bounds__(256) void prep_w_kernel(
    const float* __restrict__ conv_w, _Float16* __restrict__ Whk)
{
    int idx = blockIdx.x * blockDim.x + threadIdx.x;
    const int total = PFP * PKK;
    if (idx >= total) return;
    int k = idx % PKK;
    int f = idx / PKK;
    int s = k >> 6;          // token within window
    int d = k & 63;          // channel (padded)
    float v = 0.0f;
    if (f < PF && d < PD) v = conv_w[((size_t)f * PFS + s) * PD + d];
    Whk[idx] = (_Float16)v;
}

// ---------------- stage 3: WMMA conv + tanh + fused piecewise max pool ----------------
// grid = (3 f-tile-groups, 7 m-tiles, N), block = 32 (one wave).
// Each wave register-blocks FTW=5 filter tiles: A loaded once per K-chunk,
// reused across 5 WMMAs -> 30 v_wmma per wave, 5x less A traffic.
__global__ __launch_bounds__(32) void conv_wmma_kernel(
    const _Float16* __restrict__ Xe,   // [N][L][64]
    const _Float16* __restrict__ Whk,  // [240][192]
    const float*    __restrict__ conv_b,
    const int*      __restrict__ epos, // [N][2]
    unsigned int*   __restrict__ pools)// [N][F][3] fkey-encoded
{
    const int ftg  = blockIdx.x;       // 0..2 -> f-tiles ftg*5 .. ftg*5+4
    const int mt   = blockIdx.y;       // 0..6
    const int n    = blockIdx.z;       // sentence
    const int lane = threadIdx.x;
    const int h    = lane >> 4;        // half-wave id
    const int lid  = lane & 15;

    const int m0 = mt * 16;
    const int f0 = ftg * FTW * 16;

    // A fragment: row m = lid, time t = m0 + lid (clamped for safe loads)
    int t  = m0 + lid;
    int tc = (t < PLC - 1) ? t : (PLC - 1);            // clamp to 97
    const _Float16* arow = Xe + ((size_t)n * PL + tc) * PDP;   // 192 contiguous halves
    const _Float16* brow = Whk + (size_t)(f0 + lid) * PKK;

    v8f acc[FTW];
    #pragma unroll
    for (int u = 0; u < FTW; ++u) acc[u] = (v8f){};

    #pragma unroll
    for (int kk = 0; kk < 6; ++kk) {
        // A: element i -> k = i + 16*h + 32*kk  => one contiguous 16-half chunk
        v16h a = *(const v16h*)(arow + 32 * kk + 16 * h);
        // B: element i -> k = i + 16*h + 32*kk, col = f0 + u*16 + lid
        #pragma unroll
        for (int u = 0; u < FTW; ++u) {
            v16h b = *(const v16h*)(brow + (size_t)u * 16 * PKK + 32 * kk + 16 * h);
            acc[u] = __builtin_amdgcn_wmma_f32_16x16x32_f16(
                         false, a, false, b, (short)0, acc[u], false, false);
        }
    }

    // epilogue: tanh + piecewise max-pool via atomicMax
    const int p1 = epos[2 * n + 0];
    const int p2 = epos[2 * n + 1];

    #pragma unroll
    for (int u = 0; u < FTW; ++u) {
        const int f       = f0 + u * 16 + lid;
        const bool validf = (f < PF);
        const float bias  = validf ? conv_b[f] : 0.0f;
        unsigned int* pbase = pools + ((size_t)n * PF + (validf ? f : 0)) * 3;
        #pragma unroll
        for (int j = 0; j < 8; ++j) {
            const int tt = m0 + j + 8 * h;             // D-element row index = timestep
            if (tt < PLC && validf) {
                const float v = tanhf(acc[u][j] + bias);
                const unsigned int key = fkey(v);
                if (tt <= p1)              atomicMax(pbase + 0, key);
                if (tt >= p1 && tt <= p2)  atomicMax(pbase + 1, key);
                if (tt >= p2)              atomicMax(pbase + 2, key);
            }
        }
    }
}

// ---------------- stage 4: decode pooled keys -> feats[N][690] ----------------
__global__ __launch_bounds__(256) void decode_kernel(
    const unsigned int* __restrict__ pools, float* __restrict__ feats)
{
    int idx = blockIdx.x * blockDim.x + threadIdx.x;
    const int total = PN * F3;         // pools[n][f][s] flat == feats[n][f*3+s] flat
    if (idx >= total) return;
    feats[idx] = fdec(pools[idx]);
}

// ---------------- stage 5: per-sentence cosine attention weight ----------------
// w[n] = softmax_c(cos(feats[n], relation_mean[c]))[labels[n/8]]
__global__ __launch_bounds__(64) void att_kernel(
    const float* __restrict__ feats, const float* __restrict__ relation_mean,
    const int* __restrict__ labels, float* __restrict__ wvec)
{
    const int n   = blockIdx.x;
    const int tid = threadIdx.x;
    __shared__ float sred[64];
    __shared__ float dist[64];

    const float* fr = feats + (size_t)n * F3;
    float s = 0.0f;
    for (int k = tid; k < F3; k += 64) { float x = fr[k]; s += x * x; }
    sred[tid] = s;
    __syncthreads();
    for (int off = 32; off > 0; off >>= 1) {
        if (tid < off) sred[tid] += sred[tid + off];
        __syncthreads();
    }
    const float fn = sqrtf(sred[0]);
    __syncthreads();

    float dval = 0.0f;
    if (tid < PC) {
        const float* rm = relation_mean + (size_t)tid * F3;
        float dot = 0.0f, rs = 0.0f;
        for (int k = 0; k < F3; ++k) { float r = rm[k]; dot += fr[k] * r; rs += r * r; }
        dval = dot / fmaxf(fn * sqrtf(rs), 1e-8f);
    }
    dist[tid] = dval;
    __syncthreads();

    if (tid == 0) {
        float mx = -1e30f;
        for (int c = 0; c < PC; ++c) mx = fmaxf(mx, dist[c]);
        float se = 0.0f;
        for (int c = 0; c < PC; ++c) se += expf(dist[c] - mx);
        const int lab = labels[n >> 3];
        wvec[n] = expf(dist[lab] - mx) / se;
    }
}

// ---------------- stage 6: bag-weighted feature sum ----------------
__global__ __launch_bounds__(256) void bag_kernel(
    const float* __restrict__ feats, const float* __restrict__ wvec,
    float* __restrict__ bag_feat)
{
    const int nb  = blockIdx.x;
    const int tid = threadIdx.x;
    __shared__ float w[PBAG];
    if (tid < PBAG) w[tid] = wvec[nb * PBAG + tid];
    __syncthreads();
    if (tid == 0) {
        float s = 0.0f;
        for (int b = 0; b < PBAG; ++b) s += w[b];
        if (s < 0.001f) { for (int b = 0; b < PBAG; ++b) w[b] = 1.0f / PBAG; }
        else            { for (int b = 0; b < PBAG; ++b) w[b] /= s; }
    }
    __syncthreads();
    for (int k = tid; k < F3; k += 256) {
        float s = 0.0f;
        for (int b = 0; b < PBAG; ++b)
            s += w[b] * feats[((size_t)nb * PBAG + b) * F3 + k];
        bag_feat[(size_t)nb * F3 + k] = s;
    }
}

// ---------------- stage 7: per-bag cosine vs its relation mean ----------------
__global__ __launch_bounds__(256) void xar_kernel(
    const float* __restrict__ bag_feat, const float* __restrict__ relation_mean,
    const int* __restrict__ labels, float* __restrict__ xar)
{
    const int nb  = blockIdx.x;
    const int tid = threadIdx.x;
    const float* bf = bag_feat + (size_t)nb * F3;
    const float* rm = relation_mean + (size_t)labels[nb] * F3;
    __shared__ float sd[256], sb_[256], sr[256];
    float d = 0.0f, bn = 0.0f, rn = 0.0f;
    for (int k = tid; k < F3; k += 256) {
        float b = bf[k], r = rm[k];
        d += b * r; bn += b * b; rn += r * r;
    }
    sd[tid] = d; sb_[tid] = bn; sr[tid] = rn;
    __syncthreads();
    for (int off = 128; off > 0; off >>= 1) {
        if (tid < off) { sd[tid] += sd[tid+off]; sb_[tid] += sb_[tid+off]; sr[tid] += sr[tid+off]; }
        __syncthreads();
    }
    if (tid == 0)
        xar[nb] = sd[0] / fmaxf(sqrtf(sb_[0]) * sqrtf(sr[0]), 1e-8f);
}

// ---------------- stage 8: pair softmax + final projection ----------------
__global__ __launch_bounds__(64) void out_kernel(
    const float* __restrict__ bag_feat, const float* __restrict__ xar,
    const float* __restrict__ out_w, const float* __restrict__ out_b,
    float* __restrict__ out)
{
    const int sb  = blockIdx.x;       // 0..127
    const int tid = threadIdx.x;
    const float x0 = xar[sb * 2 + 0];
    const float x1 = xar[sb * 2 + 1];
    const float mx = fmaxf(x0, x1);
    const float e0 = expf(x0 - mx), e1 = expf(x1 - mx);
    const float inv = 1.0f / (e0 + e1);
    const float a0 = e0 * inv, a1 = e1 * inv;
    if (tid < PC) {
        const float* b0 = bag_feat + (size_t)(sb * 2 + 0) * F3;
        const float* b1 = bag_feat + (size_t)(sb * 2 + 1) * F3;
        const float* wr = out_w + (size_t)tid * F3;
        float s = out_b[tid];
        for (int k = 0; k < F3; ++k) s += (a0 * b0[k] + a1 * b1[k]) * wr[k];
        out[sb * PC + tid] = s;
    }
}

// ---------------- launch ----------------
extern "C" void kernel_launch(void* const* d_in, const int* in_sizes, int n_in,
                              void* d_out, int out_size, void* d_ws, size_t ws_size,
                              hipStream_t stream) {
    const int*   tokens  = (const int*)  d_in[0];
    const int*   pf1     = (const int*)  d_in[1];
    const int*   pf2     = (const int*)  d_in[2];
    const int*   epos    = (const int*)  d_in[3];
    const int*   labels  = (const int*)  d_in[4];
    const float* wordvec = (const float*)d_in[5];
    const float* pf1_emb = (const float*)d_in[6];
    const float* pf2_emb = (const float*)d_in[7];
    const float* conv_w  = (const float*)d_in[8];
    const float* conv_b  = (const float*)d_in[9];
    const float* rel_m   = (const float*)d_in[10];
    const float* out_w   = (const float*)d_in[11];
    const float* out_b   = (const float*)d_in[12];
    float* out = (float*)d_out;

    // workspace carve (256-byte aligned)
    char* base = (char*)d_ws;
    size_t off = 0;
    auto carve = [&](size_t bytes) {
        char* p = base + off;
        off = (off + bytes + 255) & ~(size_t)255;
        return p;
    };
    _Float16*     Xe       = (_Float16*)    carve((size_t)PN * PL * PDP * sizeof(_Float16));
    _Float16*     Whk      = (_Float16*)    carve((size_t)PFP * PKK * sizeof(_Float16));
    unsigned int* pools    = (unsigned int*)carve((size_t)PN * PF * 3 * sizeof(unsigned int));
    float*        feats    = (float*)       carve((size_t)PN * F3 * sizeof(float));
    float*        wvec     = (float*)       carve((size_t)PN * sizeof(float));
    float*        bag_feat = (float*)       carve((size_t)NBAGS * F3 * sizeof(float));
    float*        xar      = (float*)       carve((size_t)NBAGS * sizeof(float));
    (void)ws_size; (void)in_sizes; (void)n_in; (void)out_size;

    // zero pool keys (key 0 == -NaN == identity for max); capture-safe async memset
    hipMemsetAsync(pools, 0, (size_t)PN * PF * 3 * sizeof(unsigned int), stream);

    { // gather embeddings
        const int total = PN * PL * PDP;
        gather_kernel<<<(total + 255) / 256, 256, 0, stream>>>(
            tokens, pf1, pf2, wordvec, pf1_emb, pf2_emb, Xe);
    }
    { // repack conv weights
        const int total = PFP * PKK;
        prep_w_kernel<<<(total + 255) / 256, 256, 0, stream>>>(conv_w, Whk);
    }
    { // WMMA conv + fused tanh + piecewise max pool (register-blocked over filters)
        dim3 grid(PFP / 16 / FTW, (PLC + 15) / 16, PN);   // (3, 7, 2048)
        conv_wmma_kernel<<<grid, 32, 0, stream>>>(Xe, Whk, conv_b, epos, pools);
    }
    { // decode pooled keys to features
        const int total = PN * F3;
        decode_kernel<<<(total + 255) / 256, 256, 0, stream>>>(pools, feats);
    }
    // per-sentence attention weight
    att_kernel<<<PN, 64, 0, stream>>>(feats, rel_m, labels, wvec);
    // bag-weighted features
    bag_kernel<<<NBAGS, 256, 0, stream>>>(feats, wvec, bag_feat);
    // per-bag cosine score
    xar_kernel<<<NBAGS, 256, 0, stream>>>(bag_feat, rel_m, labels, xar);
    // pair softmax + output projection
    out_kernel<<<NSB, 64, 0, stream>>>(bag_feat, xar, out_w, out_b, out);
}
